// JANETNet_69939247448784
// MI455X (gfx1250) — matrix operand order repaired and checked by reading, not compiled, not run on hardware
//
#include <hip/hip_runtime.h>
#include <hip/hip_bf16.h>
#include <stdint.h>

// ---------- problem constants ----------
#define BB 64
#define SS 512
#define FF 512
#define HH 1024
#define OO 512

typedef __attribute__((ext_vector_type(16))) __bf16 v16bf;
typedef __attribute__((ext_vector_type(8)))  __bf16 v8bf;
typedef __attribute__((ext_vector_type(8)))  float  v8f;

// ---------- workspace layout (bytes) ----------
#define OFF_WF0  ((size_t)0)                 // 1536*1024 bf16 = 3145728
#define OFF_WC0  ((size_t)3145728)
#define OFF_WF1  ((size_t)6291456)           // 2048*1024 bf16 = 4194304
#define OFF_WC1  ((size_t)10485760)
#define OFF_WFC  ((size_t)14680064)          // 1024*512 bf16 = 1048576
#define OFF_H0   ((size_t)15728640)          // 2 x 64*1024 bf16 = 262144
#define OFF_H1   ((size_t)15990784)
#define OFF_C0   ((size_t)16252928)          // 64*1024 f32 = 262144
#define OFF_C1   ((size_t)16515072)
#define OFF_CTR  ((size_t)16777216)          // barrier counter (256 B slab)
#define OFF_XB   ((size_t)16777472)          // optional x in bf16: 33554432
#define WS_XBF   ((size_t)50331904)

#define HBUF (BB * HH)  // elems per h buffer

__device__ __forceinline__ float fast_sigmoid(float x) {
  return 1.f / (1.f + __expf(-x));
}
__device__ __forceinline__ float fast_tanh(float x) {
  float ax = fabsf(x);
  float e  = __expf(-2.f * ax);  // e in (0,1], no overflow
  float t  = (1.f - e) / (1.f + e);
  return copysignf(t, x);
}

// ---------- pack fp32 row-major W(K,N) into WMMA B-fragment order ----------
// element index: ((ntile*nKb + kb)*32 + lane)*16 + j
//   k = kb*32 + (j>>3)*16 + (lane>>4)*8 + (j&7) ; n = ntile*16 + (lane&15)
__global__ __launch_bounds__(256)
void janet_pack_w(const float* __restrict__ W, __bf16* __restrict__ out,
                  int K, int N) {
  size_t idx = (size_t)blockIdx.x * blockDim.x + threadIdx.x;
  size_t total = (size_t)K * N;
  if (idx >= total) return;
  int j     = (int)(idx & 15);
  int lane  = (int)((idx >> 4) & 31);
  size_t rest = idx >> 9;
  int nKb   = K >> 5;
  int kb    = (int)(rest % nKb);
  int ntile = (int)(rest / nKb);
  int k = kb * 32 + (j >> 3) * 16 + (lane >> 4) * 8 + (j & 7);
  int n = ntile * 16 + (lane & 15);
  out[idx] = (__bf16)W[(size_t)k * N + n];
}

__global__ __launch_bounds__(256)
void janet_zero(uint32_t* __restrict__ p, int n) {
  int i = blockIdx.x * blockDim.x + threadIdx.x;
  if (i < n) p[i] = 0u;
}

__global__ __launch_bounds__(256)
void janet_cvt_x(const float* __restrict__ x, __bf16* __restrict__ xb, int n) {
  int i = blockIdx.x * blockDim.x + threadIdx.x;
  if (i < n) xb[i] = (__bf16)x[i];
}

// ---------- A-fragment loaders (ISA 16-bit A 16x32 layout) ----------
// lane<16: K = {0..7, 16..23}; lane>=16: K = {8..15, 24..31} of the k-block.
__device__ __forceinline__ v16bf load_a_bf16(const __bf16* __restrict__ p) {
  v8bf lo = *(const v8bf*)(p);
  v8bf hv = *(const v8bf*)(p + 16);
  v16bf a;
#pragma unroll
  for (int j = 0; j < 8; ++j) { a[j] = lo[j]; a[j + 8] = hv[j]; }
  return a;
}
__device__ __forceinline__ v16bf load_a_f32cvt(const float* __restrict__ p) {
  v16bf a;
#pragma unroll
  for (int j = 0; j < 8; ++j) { a[j] = (__bf16)p[j]; a[j + 8] = (__bf16)p[16 + j]; }
  return a;
}

// ---------- one JANET layer for one timestep (per block: one n-tile) ----------
// A = [part0 (K0 cols, f32 or bf16) | h_prev (HH cols)]
// block: 8 waves = (mtile 0..3) x (khalf 0..1); K-halves reduced via LDS.
// All K-segment loops have compile-time trip counts (no per-iter selects).
template <int K0, bool A0F32>
__device__ __forceinline__ void layer_step(
    const float* __restrict__ a0f, const __bf16* __restrict__ a0b, size_t lda0,
    const __bf16* __restrict__ hprev,
    const __bf16* __restrict__ wfp, const __bf16* __restrict__ wcp,
    const float* __restrict__ bfv, const float* __restrict__ bcv,
    float* __restrict__ cst, __bf16* __restrict__ hout,
    float* __restrict__ sF, float* __restrict__ sC) {
  constexpr int nKb  = (K0 + HH) / 32;       // total k-blocks
  constexpr int nKbH = nKb / 2;              // per K-half
  constexpr int nKb0 = K0 / 32;              // k-blocks in part0

  const int ntile = blockIdx.x;
  const int wave  = threadIdx.x >> 5;
  const int lane  = threadIdx.x & 31;
  const int mtile = wave & 3;
  const int khalf = wave >> 2;
  const int lid   = lane & 15;
  const int hi    = lane >> 4;
  const int mrow  = mtile * 16 + lid;

  const __bf16* __restrict__ wfs = wfp + (size_t)ntile * nKb * 512 +
                                   (size_t)lane * 16;  // this n-tile's F strip
  const __bf16* __restrict__ wcs = wcp + (size_t)ntile * nKb * 512 +
                                   (size_t)lane * 16;
  const float*  __restrict__ a0fl = A0F32 ? a0f + (size_t)mrow * lda0 + hi * 8
                                          : nullptr;
  const __bf16* __restrict__ a0bl = A0F32 ? nullptr
                                          : a0b + (size_t)mrow * lda0 + hi * 8;
  const __bf16* __restrict__ hpl  = hprev + (size_t)mrow * HH + hi * 8;

  v8f accF = {0.f, 0.f, 0.f, 0.f, 0.f, 0.f, 0.f, 0.f};
  v8f accC = {0.f, 0.f, 0.f, 0.f, 0.f, 0.f, 0.f, 0.f};

  auto mm = [&](v16bf a, int kb) {
    const size_t woff = (size_t)kb * 512;                // 32 lanes * 16 elems
    v16bf bF = *(const v16bf*)(wfs + woff);
    v16bf bC = *(const v16bf*)(wcs + woff);
    // unconditional prefetch 4 k-blocks ahead (<=4KB over-run stays in ws)
    const size_t pf = woff + 4 * 512;
    __builtin_prefetch(wfs + pf, 0, 1);
    __builtin_prefetch(wcs + pf, 0, 1);
    // two independent accumulation chains -> WMMA latency hiding
    accF = __builtin_amdgcn_wmma_f32_16x16x32_bf16(false, a, false, bF,
                                                   (short)0, accF, false, false);
    accC = __builtin_amdgcn_wmma_f32_16x16x32_bf16(false, a, false, bC,
                                                   (short)0, accC, false, false);
  };

  if (khalf == 0) {
    constexpr int n0 = (nKb0 < nKbH) ? nKb0 : nKbH;  // part0 blocks, lower half
#pragma unroll 2
    for (int kb = 0; kb < n0; ++kb) {
      v16bf a;
      if constexpr (A0F32) a = load_a_f32cvt(a0fl + kb * 32);
      else                 a = load_a_bf16(a0bl + kb * 32);
      mm(a, kb);
    }
#pragma unroll 2
    for (int kb = n0; kb < nKbH; ++kb) {             // h_prev blocks, lower half
      mm(load_a_bf16(hpl + kb * 32 - K0), kb);
    }
  } else {
    constexpr int n1 = (nKb0 > nKbH) ? nKb0 : nKbH;  // part0 end in upper half
#pragma unroll 2
    for (int kb = nKbH; kb < n1; ++kb) {             // part0 leftovers (layer0: none)
      v16bf a;
      if constexpr (A0F32) a = load_a_f32cvt(a0fl + kb * 32);
      else                 a = load_a_bf16(a0bl + kb * 32);
      mm(a, kb);
    }
#pragma unroll 2
    for (int kb = n1; kb < nKb; ++kb) {              // h_prev blocks, upper half
      mm(load_a_bf16(hpl + kb * 32 - K0), kb);
    }
  }

  if (khalf) {
#pragma unroll
    for (int r = 0; r < 8; ++r) {
      sF[(mtile * 16 + r + 8 * hi) * 16 + lid] = accF[r];
      sC[(mtile * 16 + r + 8 * hi) * 16 + lid] = accC[r];
    }
  }
  __syncthreads();
  if (!khalf) {
    const int n = ntile * 16 + lid;
    const float biasF = bfv[n];
    const float biasC = bcv[n];
#pragma unroll
    for (int r = 0; r < 8; ++r) {
      const int m = mtile * 16 + r + 8 * hi;
      const float gF = accF[r] + sF[(mtile * 16 + r + 8 * hi) * 16 + lid] + biasF;
      const float gC = accC[r] + sC[(mtile * 16 + r + 8 * hi) * 16 + lid] + biasC;
      const float f  = fast_sigmoid(gF);
      const float ct = fast_tanh(gC);
      const size_t off = (size_t)m * HH + n;
      const float c = f * cst[off] + (1.f - f) * ct;
      cst[off]  = c;
      hout[off] = (__bf16)fast_tanh(c);
    }
  }
}

// ---------- fc tile: out[mtile,ntile] = h(64x1024) @ Wfc + bfc ----------
__device__ __forceinline__ void fc_tile(int mtile, int ntile,
                                        const __bf16* __restrict__ h,
                                        const __bf16* __restrict__ wp,
                                        const float* __restrict__ bias,
                                        float* __restrict__ out, int lane) {
  constexpr int nKb = HH / 32;  // 32
  const int lid  = lane & 15;
  const int hi   = lane >> 4;
  const int mrow = mtile * 16 + lid;
  const __bf16* __restrict__ wps = wp + (size_t)ntile * nKb * 512 +
                                   (size_t)lane * 16;
  const __bf16* __restrict__ hl  = h + (size_t)mrow * HH + hi * 8;
  v8f acc = {0.f, 0.f, 0.f, 0.f, 0.f, 0.f, 0.f, 0.f};
#pragma unroll 2
  for (int kb = 0; kb < nKb; ++kb) {
    v16bf a = load_a_bf16(hl + kb * 32);
    v16bf b = *(const v16bf*)(wps + (size_t)kb * 512);
    acc = __builtin_amdgcn_wmma_f32_16x16x32_bf16(false, a, false, b,
                                                  (short)0, acc, false, false);
  }
  const int n = ntile * 16 + lid;
  const float bv = bias[n];
#pragma unroll
  for (int r = 0; r < 8; ++r) {
    const int m = mtile * 16 + r + 8 * hi;
    out[(size_t)m * OO + n] = acc[r] + bv;
  }
}

// ---------- device-wide barrier (64 co-resident blocks, generation count) ----
__device__ __forceinline__ void grid_barrier(unsigned* __restrict__ ctr,
                                             unsigned expected) {
  __syncthreads();
  if (threadIdx.x == 0) {
    __threadfence();  // release: make this block's stores device-visible
    __hip_atomic_fetch_add(ctr, 1u, __ATOMIC_RELEASE, __HIP_MEMORY_SCOPE_AGENT);
#pragma unroll 1
    while (__hip_atomic_load(ctr, __ATOMIC_ACQUIRE, __HIP_MEMORY_SCOPE_AGENT) <
           expected) {
      __builtin_amdgcn_s_sleep(2);
    }
  }
  __syncthreads();
  __threadfence();  // acquire for every wave: drop stale cached lines
}

// ---------- persistent kernel: all 512 steps + final FC ----------
// grid: 64 blocks (one n-tile each) x 256 threads. Blocks are guaranteed
// co-resident (64 WGs, 8KB LDS, modest VGPRs) -> spin barrier is safe.
template <bool XBF16>
__global__ __launch_bounds__(256)
void janet_persistent(const float* __restrict__ xf,
                      const __bf16* __restrict__ xb,
                      const __bf16* __restrict__ wf0p,
                      const __bf16* __restrict__ wc0p,
                      const __bf16* __restrict__ wf1p,
                      const __bf16* __restrict__ wc1p,
                      const __bf16* __restrict__ wfcp,
                      const float* __restrict__ bf0,
                      const float* __restrict__ bc0,
                      const float* __restrict__ bf1,
                      const float* __restrict__ bc1,
                      const float* __restrict__ bfc,
                      float* __restrict__ c0, float* __restrict__ c1,
                      __bf16* __restrict__ h0b, __bf16* __restrict__ h1b,
                      float* __restrict__ out, unsigned* __restrict__ ctr) {
  __shared__ float sF[4 * 16 * 16];
  __shared__ float sC[4 * 16 * 16];
  const unsigned nb = gridDim.x;
  unsigned bar = 0;
  int cur = 0;
#pragma unroll 1
  for (int t = 0; t < SS; ++t) {
    __bf16* h0p = h0b + (size_t)cur * HBUF;
    __bf16* h0n = h0b + (size_t)(1 - cur) * HBUF;
    __bf16* h1p = h1b + (size_t)cur * HBUF;
    __bf16* h1n = h1b + (size_t)(1 - cur) * HBUF;
    // layer 0: A = [x_t | h0_prev]
    if constexpr (XBF16) {
      layer_step<FF, false>(nullptr, xb + (size_t)t * FF, (size_t)SS * FF,
                            h0p, wf0p, wc0p, bf0, bc0, c0, h0n, sF, sC);
    } else {
      layer_step<FF, true>(xf + (size_t)t * FF, nullptr, (size_t)SS * FF,
                           h0p, wf0p, wc0p, bf0, bc0, c0, h0n, sF, sC);
    }
    grid_barrier(ctr, (++bar) * nb);
    // layer 1: A = [h0_new | h1_prev]
    layer_step<HH, false>(nullptr, h0n, (size_t)HH,
                          h1p, wf1p, wc1p, bf1, bc1, c1, h1n, sF, sC);
    grid_barrier(ctr, (++bar) * nb);
    cur ^= 1;
  }
  // final FC on last hidden state (HORIZON == 1): 128 wave-tiles
  const int w = blockIdx.x * 8 + (threadIdx.x >> 5);
  if (w < (BB / 16) * (OO / 16)) {
    fc_tile(w & 3, w >> 2, h1b + (size_t)cur * HBUF, wfcp, bfc, out,
            threadIdx.x & 31);
  }
}

// ---------- host driver ----------
extern "C" void kernel_launch(void* const* d_in, const int* in_sizes, int n_in,
                              void* d_out, int out_size, void* d_ws, size_t ws_size,
                              hipStream_t stream) {
  const float* x   = (const float*)d_in[0];
  const float* Wf0 = (const float*)d_in[1];
  const float* bf0 = (const float*)d_in[2];
  const float* Wc0 = (const float*)d_in[3];
  const float* bc0 = (const float*)d_in[4];
  const float* Wf1 = (const float*)d_in[5];
  const float* bf1 = (const float*)d_in[6];
  const float* Wc1 = (const float*)d_in[7];
  const float* bc1 = (const float*)d_in[8];
  const float* Wfc = (const float*)d_in[9];
  const float* bfc = (const float*)d_in[10];
  float* out = (float*)d_out;

  char* ws = (char*)d_ws;
  __bf16*   wf0p = (__bf16*)(ws + OFF_WF0);
  __bf16*   wc0p = (__bf16*)(ws + OFF_WC0);
  __bf16*   wf1p = (__bf16*)(ws + OFF_WF1);
  __bf16*   wc1p = (__bf16*)(ws + OFF_WC1);
  __bf16*   wfcp = (__bf16*)(ws + OFF_WFC);
  __bf16*   h0b  = (__bf16*)(ws + OFF_H0);
  __bf16*   h1b  = (__bf16*)(ws + OFF_H1);
  float*    c0   = (float*)(ws + OFF_C0);
  float*    c1   = (float*)(ws + OFF_C1);
  unsigned* ctr  = (unsigned*)(ws + OFF_CTR);
  __bf16*   xb   = (__bf16*)(ws + OFF_XB);

  const bool use_xb = (ws_size >= WS_XBF);  // fixed per process -> deterministic

  // pack weights into bf16 WMMA-B fragment order (idempotent, every call)
  {
    int e0 = (FF + HH) * HH;  // 1572864
    int e1 = (HH + HH) * HH;  // 2097152
    int e2 = HH * OO;         // 524288
    janet_pack_w<<<(e0 + 255) / 256, 256, 0, stream>>>(Wf0, wf0p, FF + HH, HH);
    janet_pack_w<<<(e0 + 255) / 256, 256, 0, stream>>>(Wc0, wc0p, FF + HH, HH);
    janet_pack_w<<<(e1 + 255) / 256, 256, 0, stream>>>(Wf1, wf1p, HH + HH, HH);
    janet_pack_w<<<(e1 + 255) / 256, 256, 0, stream>>>(Wc1, wc1p, HH + HH, HH);
    janet_pack_w<<<(e2 + 255) / 256, 256, 0, stream>>>(Wfc, wfcp, HH, OO);
  }
  // zero h buffers, c state and the barrier counter (contiguous region)
  {
    int words = (int)((OFF_XB - OFF_H0) / 4);
    janet_zero<<<(words + 255) / 256, 256, 0, stream>>>(
        (uint32_t*)(ws + OFF_H0), words);
  }
  if (use_xb) {
    int n = BB * SS * FF;  // 16777216
    janet_cvt_x<<<(n + 255) / 256, 256, 0, stream>>>(x, xb, n);
    janet_persistent<true><<<HH / 16, 256, 0, stream>>>(
        nullptr, xb, wf0p, wc0p, wf1p, wc1p, wfcp,
        bf0, bc0, bf1, bc1, bfc, c0, c1, h0b, h1b, out, ctr);
  } else {
    janet_persistent<false><<<HH / 16, 256, 0, stream>>>(
        x, nullptr, wf0p, wc0p, wf1p, wc1p, wfcp,
        bf0, bc0, bf1, bc1, bfc, c0, c1, h0b, h1b, out, ctr);
  }
  (void)in_sizes; (void)n_in; (void)out_size; (void)ws_size;
}
